// RealImagGQA_17978733101524
// MI455X (gfx1250) — matrix operand order repaired and checked by reading, not compiled, not run on hardware
//
#include <hip/hip_runtime.h>
#include <stdint.h>

#define B_    2
#define S_    2048
#define HID_  2048
#define NH_   16
#define NKV_  4
#define HD_   128
#define GROUPS_ (NH_ / NKV_)
#define KVW_  (NKV_ * HD_)   // 512

#if __has_builtin(__builtin_amdgcn_global_load_async_to_lds_b128) && \
    __has_builtin(__builtin_amdgcn_s_wait_asynccnt)
#define CDNA5_ASYNC_LDS 1
#else
#define CDNA5_ASYNC_LDS 0
#endif

typedef __attribute__((ext_vector_type(16))) __bf16 v16bf;
typedef __attribute__((ext_vector_type(8)))  __bf16 v8bf;
typedef __attribute__((ext_vector_type(8)))  float  v8f;
typedef __attribute__((ext_vector_type(4)))  int    v4i;

// ---------- scalar converters (round-to-nearest-even) ----------
__device__ __forceinline__ __bf16 f2bf(float f) {
  union { float f; uint32_t u; } v; v.f = f;
  uint32_t r = (v.u + 0x7FFFu + ((v.u >> 16) & 1u)) >> 16;
  unsigned short h = (unsigned short)r;
  __bf16 o; __builtin_memcpy(&o, &h, 2);
  return o;
}
__device__ __forceinline__ float bf2f(__bf16 b) {
  unsigned short h; __builtin_memcpy(&h, &b, 2);
  union { uint32_t u; float f; } v; v.u = ((uint32_t)h) << 16;
  return v.f;
}

// ---------- WMMA wrapper ----------
__device__ __forceinline__ v8f wmma_bf16(v16bf a, v16bf b, v8f c) {
  return __builtin_amdgcn_wmma_f32_16x16x32_bf16(
      /*neg_a=*/false, a, /*neg_b=*/false, b,
      /*c_mod=*/(short)0, c, /*reuse_a=*/false, /*reuse_b=*/false);
}

// A-fragment (16x32 bf16), stride-32-elements source (works for global or LDS):
//   elems 0..15 = K[half*16 .. +15]
__device__ __forceinline__ v16bf load_a_frag(const __bf16* p /* row base + half*16 */) {
  v8bf lo = *(const v8bf*)(p);
  v8bf hi = *(const v8bf*)(p + 8);
  v16bf a;
#pragma unroll
  for (int i = 0; i < 8; ++i) { a[i] = lo[i]; a[i + 8] = hi[i]; }
  return a;
}

// ---------- LDS staging primitive: 16B global -> LDS ----------
__device__ __forceinline__ void stage_b128(__bf16* dst_lds, const __bf16* src_glb) {
#if CDNA5_ASYNC_LDS
  __builtin_amdgcn_global_load_async_to_lds_b128(
      (__attribute__((address_space(1))) v4i*)src_glb,   // C-cast: drops const + addrspacecast
      (__attribute__((address_space(3))) v4i*)dst_lds,
      /*offset=*/0, /*cpol=*/0);
#else
  *(v8bf*)dst_lds = *(const v8bf*)src_glb;
#endif
}
__device__ __forceinline__ void stage_commit() {
#if CDNA5_ASYNC_LDS
  __builtin_amdgcn_s_wait_asynccnt(0);   // my async LDS writes are done
#endif
  __syncthreads();                        // everyone's writes visible
}

// ---------- elementwise helpers ----------
__global__ void __launch_bounds__(256)
f32_to_bf16_kernel(const float* __restrict__ in, __bf16* __restrict__ out, int n) {
  int i = blockIdx.x * blockDim.x + threadIdx.x;
  if (i < n) out[i] = f2bf(in[i]);
}

// W[k,n] f32 -> Wt[n,k] bf16 (optionally negated)
__global__ void __launch_bounds__(256)
transposeW_kernel(const float* __restrict__ in, __bf16* __restrict__ out, int K, int N, int neg) {
  int i = blockIdx.x * blockDim.x + threadIdx.x;
  if (i >= K * N) return;
  int k = i / N, n = i - k * N;
  float v = in[i];
  out[(size_t)n * K + k] = f2bf(neg ? -v : v);
}

// in-place complex RoPE on [B,S,W] bf16 pair (W = heads*HD), freq index = (c%HD)%64
__global__ void __launch_bounds__(256)
rope_kernel(__bf16* __restrict__ xr, __bf16* __restrict__ xi, int W, int total) {
  int i = blockIdx.x * blockDim.x + threadIdx.x;
  if (i >= total) return;
  int c = i % W;
  int s = (i / W) % S_;
  int fidx = (c % HD_) & 63;
  float inv = __expf(-(float)fidx * 0.14391156831212787f);  // 10000^(-fidx/64)
  float ang = (float)s * inv;
  float sn, cs;
  __sincosf(ang, &sn, &cs);
  float r = bf2f(xr[i]), im = bf2f(xi[i]);
  xr[i] = f2bf(r * cs - im * sn);
  xi[i] = f2bf(r * sn + im * cs);
}

// v [B,S,KVW] bf16 -> vt [B,NKV,HD,S] bf16
__global__ void __launch_bounds__(256)
transposeV_kernel(const __bf16* __restrict__ v, __bf16* __restrict__ vt, int n) {
  int i = blockIdx.x * blockDim.x + threadIdx.x;
  if (i >= n) return;
  int c = i % KVW_;
  int s = (i / KVW_) % S_;
  int b = i / (KVW_ * S_);
  int kv = c / HD_, d = c - kv * HD_;
  vt[((size_t)(b * NKV_ + kv) * HD_ + d) * S_ + s] = v[i];
}

// ---------- complex GEMM, 2x2 register-blocked, LDS-staged A ----------
// out_r = Ar@Br + Ai@(-Bi) + biasR ; out_i = Ar@Bi + Ai@Br + biasI
// A: [M,K] bf16. BrT/BiT/BinT: [N,K] bf16 (transposed; BinT = -Bi).
// Block = 4 waves sharing one 32-row A tile (staged via async-to-LDS, double
// buffered); each wave owns a 32-col B slice -> 32x32 C tile per wave.
template <bool OUT_BF16>
__global__ void __launch_bounds__(128)
cgemm_kernel(const __bf16* __restrict__ Ar, const __bf16* __restrict__ Ai,
             const __bf16* __restrict__ BrT, const __bf16* __restrict__ BiT,
             const __bf16* __restrict__ BinT,
             const float* __restrict__ biasR, const float* __restrict__ biasI,
             void* __restrict__ outR_, void* __restrict__ outI_,
             int N, int K) {
  __shared__ __bf16 stage[2][2][32 * 32];  // [buf][r/i][row*32 + k]

  const int tid = threadIdx.x;
  const int lane = tid & 31;
  const int wave = tid >> 5;
  const int half = lane >> 4, l16 = lane & 15;
  const int m0 = blockIdx.x * 32;
  const int n0 = (blockIdx.y * 4 + wave) * 32;  // grid sized so n0+32 <= N always

  // cooperative staging: each of 128 threads copies one 16B chunk of Ar and Ai
  const int srow = tid >> 2;           // 0..31
  const int scol = (tid & 3) * 8;      // 0,8,16,24 (bf16 elements)
  const __bf16* gAr = Ar + (size_t)(m0 + srow) * K + scol;
  const __bf16* gAi = Ai + (size_t)(m0 + srow) * K + scol;
  const int soff = srow * 32 + scol;

  const __bf16* bR[2], *bI[2], *bN[2];
#pragma unroll
  for (int t = 0; t < 2; ++t) {
    bR[t] = BrT + (size_t)(n0 + t * 16 + l16) * K + half * 16;
    bI[t] = BiT + (size_t)(n0 + t * 16 + l16) * K + half * 16;
    bN[t] = BinT + (size_t)(n0 + t * 16 + l16) * K + half * 16;
  }

  // prologue: stage k-step 0 into buffer 0
  stage_b128(&stage[0][0][soff], gAr);
  stage_b128(&stage[0][1][soff], gAi);

  v8f accR[2][2] = {};  // [mi][ni]
  v8f accI[2][2] = {};
  const int nk = K / 32;
  for (int kt = 0; kt < nk; ++kt) {
    const int buf = kt & 1;
    const int k0 = kt * 32;
    stage_commit();  // s_wait_asynccnt 0 + barrier: buffer `buf` ready
    if (kt + 1 < nk) {  // prefetch next tile into the other buffer
      stage_b128(&stage[buf ^ 1][0][soff], gAr + k0 + 32);
      stage_b128(&stage[buf ^ 1][1][soff], gAi + k0 + 32);
    }
    __builtin_prefetch(bR[0] + k0 + 256, 0, 1);

    v16bf Af_r[2], Af_i[2];
#pragma unroll
    for (int t = 0; t < 2; ++t) {
      Af_r[t] = load_a_frag(&stage[buf][0][(t * 16 + l16) * 32 + half * 16]);
      Af_i[t] = load_a_frag(&stage[buf][1][(t * 16 + l16) * 32 + half * 16]);
    }
    v16bf Bf_r[2], Bf_i[2], Bf_n[2];
#pragma unroll
    for (int t = 0; t < 2; ++t) {
      Bf_r[t] = *(const v16bf*)(bR[t] + k0);
      Bf_i[t] = *(const v16bf*)(bI[t] + k0);
      Bf_n[t] = *(const v16bf*)(bN[t] + k0);
    }
#pragma unroll
    for (int mi = 0; mi < 2; ++mi) {
#pragma unroll
      for (int ni = 0; ni < 2; ++ni) {
        accR[mi][ni] = wmma_bf16(Af_r[mi], Bf_r[ni], accR[mi][ni]);
        accR[mi][ni] = wmma_bf16(Af_i[mi], Bf_n[ni], accR[mi][ni]);
        accI[mi][ni] = wmma_bf16(Af_r[mi], Bf_i[ni], accI[mi][ni]);
        accI[mi][ni] = wmma_bf16(Af_i[mi], Bf_r[ni], accI[mi][ni]);
      }
    }
  }

#pragma unroll
  for (int mi = 0; mi < 2; ++mi) {
#pragma unroll
    for (int ni = 0; ni < 2; ++ni) {
      const int col = n0 + ni * 16 + l16;
      const float brv = biasR[col], biv = biasI[col];
#pragma unroll
      for (int j = 0; j < 8; ++j) {
        size_t off = (size_t)(m0 + mi * 16 + half * 8 + j) * N + col;
        if constexpr (OUT_BF16) {
          ((__bf16*)outR_)[off] = f2bf(accR[mi][ni][j] + brv);
          ((__bf16*)outI_)[off] = f2bf(accI[mi][ni][j] + biv);
        } else {
          ((float*)outR_)[off] = accR[mi][ni][j] + brv;
          ((float*)outI_)[off] = accI[mi][ni][j] + biv;
        }
      }
    }
  }
}

// ---------- attention scores: real(q . conj(k)) * scale, causal mask ----------
// Each wave: one 16-row q tile x two 16-col k tiles (q fragments reused).
__global__ void __launch_bounds__(128)
scores_kernel(const __bf16* __restrict__ qR, const __bf16* __restrict__ qI,
              const __bf16* __restrict__ kR, const __bf16* __restrict__ kI,
              float* __restrict__ attn) {
  const int lane = threadIdx.x & 31;
  const int wave = threadIdx.x >> 5;
  const int half = lane >> 4, l16 = lane & 15;
  const int qt = blockIdx.x;
  const int bh = blockIdx.z;
  const int b = bh / NH_, h = bh - b * NH_;
  const int kvh = h / GROUPS_;
  const int q0 = qt * 16;
  const int k0s = (blockIdx.y * 4 + wave) * 32;  // two tiles: k0s, k0s+16
  float* aP = attn + (size_t)bh * S_ * S_;

  if (k0s > q0 + 15) {  // both tiles fully masked (wave-uniform)
#pragma unroll
    for (int t = 0; t < 2; ++t)
#pragma unroll
      for (int j = 0; j < 8; ++j)
        aP[(size_t)(q0 + half * 8 + j) * S_ + (k0s + t * 16 + l16)] = -1e9f;
    return;
  }

  const __bf16* qRp = qR + (size_t)b * S_ * HID_ + (size_t)(q0 + l16) * HID_ + h * HD_ + half * 16;
  const __bf16* qIp = qI + (size_t)b * S_ * HID_ + (size_t)(q0 + l16) * HID_ + h * HD_ + half * 16;
  const __bf16* kRp[2], *kIp[2];
#pragma unroll
  for (int t = 0; t < 2; ++t) {
    kRp[t] = kR + (size_t)b * S_ * KVW_ + (size_t)(k0s + t * 16 + l16) * KVW_ + kvh * HD_ + half * 16;
    kIp[t] = kI + (size_t)b * S_ * KVW_ + (size_t)(k0s + t * 16 + l16) * KVW_ + kvh * HD_ + half * 16;
  }

  v8f acc[2] = {};
#pragma unroll
  for (int d0 = 0; d0 < HD_; d0 += 32) {
    v16bf a_r = load_a_frag(qRp + d0);
    v16bf a_i = load_a_frag(qIp + d0);
#pragma unroll
    for (int t = 0; t < 2; ++t) {
      v16bf b_r = *(const v16bf*)(kRp[t] + d0);
      v16bf b_i = *(const v16bf*)(kIp[t] + d0);
      acc[t] = wmma_bf16(a_r, b_r, acc[t]);
      acc[t] = wmma_bf16(a_i, b_i, acc[t]);
    }
  }

  const float scale = 0.08838834764831845f;  // 1/sqrt(128)
#pragma unroll
  for (int t = 0; t < 2; ++t) {
#pragma unroll
    for (int j = 0; j < 8; ++j) {
      int qrow = q0 + half * 8 + j;
      int kcol = k0s + t * 16 + l16;
      float v = (kcol <= qrow) ? acc[t][j] * scale : -1e9f;
      aP[(size_t)qrow * S_ + kcol] = v;
    }
  }
}

// ---------- row softmax over S columns, in-place ----------
__global__ void __launch_bounds__(256)
softmax_kernel(float* __restrict__ attn) {
  __shared__ float red[256];
  float* p = attn + (size_t)blockIdx.x * S_;
  const int t = threadIdx.x;

  float m = -3.4e38f;
  for (int c = t; c < S_; c += 256) m = fmaxf(m, p[c]);
  red[t] = m; __syncthreads();
  for (int s = 128; s > 0; s >>= 1) { if (t < s) red[t] = fmaxf(red[t], red[t + s]); __syncthreads(); }
  m = red[0]; __syncthreads();

  float sum = 0.f;
  for (int c = t; c < S_; c += 256) { float e = __expf(p[c] - m); p[c] = e; sum += e; }
  red[t] = sum; __syncthreads();
  for (int s = 128; s > 0; s >>= 1) { if (t < s) red[t] += red[t + s]; __syncthreads(); }
  float inv = 1.0f / red[0];

  for (int c = t; c < S_; c += 256) p[c] *= inv;
}

// ---------- ctx = attn @ v (r and i), causal K bound ----------
// Each wave: one 16-row q tile x two 16-wide d slices (attn fragment reused 4x).
// attn: [B*NH, S, S] f32; vt: [B,NKV,HD,S] bf16; ctx: [B,S,HID] bf16
__global__ void __launch_bounds__(128)
attnv_kernel(const float* __restrict__ attn,
             const __bf16* __restrict__ vtR, const __bf16* __restrict__ vtI,
             __bf16* __restrict__ ctxR, __bf16* __restrict__ ctxI) {
  const int lane = threadIdx.x & 31;
  const int wave = threadIdx.x >> 5;
  const int half = lane >> 4, l16 = lane & 15;
  const int qt = blockIdx.x;
  const int bh = blockIdx.y;
  const int b = bh / NH_, h = bh - b * NH_;
  const int kvh = h / GROUPS_;
  const int q0 = qt * 16;
  const int d0n = wave * 32;  // two d tiles: d0n, d0n+16   (4 waves cover HD=128)

  const float* aP = attn + (size_t)bh * S_ * S_ + (size_t)(q0 + l16) * S_ + half * 16;
  const __bf16* vR[2], *vI[2];
#pragma unroll
  for (int t = 0; t < 2; ++t) {
    vR[t] = vtR + ((size_t)(b * NKV_ + kvh) * HD_ + d0n + t * 16 + l16) * S_ + half * 16;
    vI[t] = vtI + ((size_t)(b * NKV_ + kvh) * HD_ + d0n + t * 16 + l16) * S_ + half * 16;
  }

  v8f accR[2] = {}, accI[2] = {};
  const int kend = q0 + 16;  // keys > q0+15 have attn == 0 exactly
  for (int k0 = 0; k0 < kend; k0 += 32) {
    v8f lo = *(const v8f*)(aP + k0);
    v8f hi = *(const v8f*)(aP + k0 + 8);
    v16bf a;
#pragma unroll
    for (int i = 0; i < 8; ++i) { a[i] = f2bf(lo[i]); a[i + 8] = f2bf(hi[i]); }
#pragma unroll
    for (int t = 0; t < 2; ++t) {
      v16bf br = *(const v16bf*)(vR[t] + k0);
      v16bf bi = *(const v16bf*)(vI[t] + k0);
      accR[t] = wmma_bf16(a, br, accR[t]);
      accI[t] = wmma_bf16(a, bi, accI[t]);
    }
  }

#pragma unroll
  for (int t = 0; t < 2; ++t) {
#pragma unroll
    for (int j = 0; j < 8; ++j) {
      size_t off = (size_t)b * S_ * HID_ + (size_t)(q0 + half * 8 + j) * HID_ +
                   h * HD_ + d0n + t * 16 + l16;
      ctxR[off] = f2bf(accR[t][j]);
      ctxI[off] = f2bf(accI[t][j]);
    }
  }
}

// ---------------------------------------------------------------------------
extern "C" void kernel_launch(void* const* d_in, const int* in_sizes, int n_in,
                              void* d_out, int out_size, void* d_ws, size_t ws_size,
                              hipStream_t stream) {
  (void)in_sizes; (void)n_in; (void)out_size; (void)ws_size;
  const float* x_r  = (const float*)d_in[0];
  const float* x_i  = (const float*)d_in[1];
  const float* Wq_r = (const float*)d_in[2];
  const float* Wq_i = (const float*)d_in[3];
  const float* bq_r = (const float*)d_in[4];
  const float* bq_i = (const float*)d_in[5];
  const float* Wk_r = (const float*)d_in[6];
  const float* Wk_i = (const float*)d_in[7];
  const float* bk_r = (const float*)d_in[8];
  const float* bk_i = (const float*)d_in[9];
  const float* Wv_r = (const float*)d_in[10];
  const float* Wv_i = (const float*)d_in[11];
  const float* bv_r = (const float*)d_in[12];
  const float* bv_i = (const float*)d_in[13];
  const float* Wo_r = (const float*)d_in[14];
  const float* Wo_i = (const float*)d_in[15];
  const float* bo_r = (const float*)d_in[16];
  const float* bo_i = (const float*)d_in[17];

  const size_t SZ_X = (size_t)B_ * S_ * HID_;   // 8,388,608
  const size_t SZ_K = (size_t)B_ * S_ * KVW_;   // 2,097,152
  const size_t W_Q  = (size_t)HID_ * HID_;
  const size_t W_K  = (size_t)HID_ * KVW_;

  char* ws = (char*)d_ws;
  size_t off = 0;
  auto carve = [&](size_t bytes) -> void* {
    void* p = (void*)(ws + off);
    off += (bytes + 255) & ~(size_t)255;
    return p;
  };
  __bf16* xr16 = (__bf16*)carve(SZ_X * 2);
  __bf16* xi16 = (__bf16*)carve(SZ_X * 2);
  __bf16* wqrt = (__bf16*)carve(W_Q * 2);
  __bf16* wqit = (__bf16*)carve(W_Q * 2);
  __bf16* wqin = (__bf16*)carve(W_Q * 2);   // -Wq_i^T
  __bf16* wkrt = (__bf16*)carve(W_K * 2);
  __bf16* wkit = (__bf16*)carve(W_K * 2);
  __bf16* wkin = (__bf16*)carve(W_K * 2);
  __bf16* wvrt = (__bf16*)carve(W_K * 2);
  __bf16* wvit = (__bf16*)carve(W_K * 2);
  __bf16* wvin = (__bf16*)carve(W_K * 2);
  __bf16* wort = (__bf16*)carve(W_Q * 2);
  __bf16* woit = (__bf16*)carve(W_Q * 2);
  __bf16* woin = (__bf16*)carve(W_Q * 2);
  __bf16* qr   = (__bf16*)carve(SZ_X * 2);
  __bf16* qi   = (__bf16*)carve(SZ_X * 2);
  __bf16* kr   = (__bf16*)carve(SZ_K * 2);
  __bf16* ki   = (__bf16*)carve(SZ_K * 2);
  __bf16* vr   = (__bf16*)carve(SZ_K * 2);
  __bf16* vi   = (__bf16*)carve(SZ_K * 2);
  __bf16* vtr  = (__bf16*)carve(SZ_K * 2);
  __bf16* vti  = (__bf16*)carve(SZ_K * 2);
  __bf16* ctxr = (__bf16*)carve(SZ_X * 2);
  __bf16* ctxi = (__bf16*)carve(SZ_X * 2);

  const int T = 256;
  // 1) activations -> bf16
  f32_to_bf16_kernel<<<(int)((SZ_X + T - 1) / T), T, 0, stream>>>(x_r, xr16, (int)SZ_X);
  f32_to_bf16_kernel<<<(int)((SZ_X + T - 1) / T), T, 0, stream>>>(x_i, xi16, (int)SZ_X);
  // 2) weights -> bf16 transposed [N,K] (+ negated imag copies)
  const int gWQ = (int)((W_Q + T - 1) / T), gWK = (int)((W_K + T - 1) / T);
  transposeW_kernel<<<gWQ, T, 0, stream>>>(Wq_r, wqrt, HID_, HID_, 0);
  transposeW_kernel<<<gWQ, T, 0, stream>>>(Wq_i, wqit, HID_, HID_, 0);
  transposeW_kernel<<<gWQ, T, 0, stream>>>(Wq_i, wqin, HID_, HID_, 1);
  transposeW_kernel<<<gWK, T, 0, stream>>>(Wk_r, wkrt, HID_, KVW_, 0);
  transposeW_kernel<<<gWK, T, 0, stream>>>(Wk_i, wkit, HID_, KVW_, 0);
  transposeW_kernel<<<gWK, T, 0, stream>>>(Wk_i, wkin, HID_, KVW_, 1);
  transposeW_kernel<<<gWK, T, 0, stream>>>(Wv_r, wvrt, HID_, KVW_, 0);
  transposeW_kernel<<<gWK, T, 0, stream>>>(Wv_i, wvit, HID_, KVW_, 0);
  transposeW_kernel<<<gWK, T, 0, stream>>>(Wv_i, wvin, HID_, KVW_, 1);
  transposeW_kernel<<<gWQ, T, 0, stream>>>(Wo_r, wort, HID_, HID_, 0);
  transposeW_kernel<<<gWQ, T, 0, stream>>>(Wo_i, woit, HID_, HID_, 0);
  transposeW_kernel<<<gWQ, T, 0, stream>>>(Wo_i, woin, HID_, HID_, 1);
  // 3) complex QKV projections (WMMA, 32x32 per wave, LDS-staged A)
  dim3 gq(B_ * S_ / 32, HID_ / 32 / 4);  // (128, 16)
  dim3 gk(B_ * S_ / 32, KVW_ / 32 / 4);  // (128, 4)
  cgemm_kernel<true><<<gq, 128, 0, stream>>>(xr16, xi16, wqrt, wqit, wqin, bq_r, bq_i, qr, qi, HID_, HID_);
  cgemm_kernel<true><<<gk, 128, 0, stream>>>(xr16, xi16, wkrt, wkit, wkin, bk_r, bk_i, kr, ki, KVW_, HID_);
  cgemm_kernel<true><<<gk, 128, 0, stream>>>(xr16, xi16, wvrt, wvit, wvin, bv_r, bv_i, vr, vi, KVW_, HID_);
  // 4) RoPE in place on q, k
  rope_kernel<<<(int)((SZ_X + T - 1) / T), T, 0, stream>>>(qr, qi, HID_, (int)SZ_X);
  rope_kernel<<<(int)((SZ_K + T - 1) / T), T, 0, stream>>>(kr, ki, KVW_, (int)SZ_K);
  // 5) transpose v -> [B,NKV,HD,S]
  transposeV_kernel<<<(int)((SZ_K + T - 1) / T), T, 0, stream>>>(vr, vtr, (int)SZ_K);
  transposeV_kernel<<<(int)((SZ_K + T - 1) / T), T, 0, stream>>>(vi, vti, (int)SZ_K);
  // 6) scores (WMMA) -> attn region of d_out, then softmax in place
  float* outR = (float*)d_out;
  float* outI = outR + SZ_X;
  float* attn = outI + SZ_X;
  dim3 gs(S_ / 16, S_ / 32 / 4, B_ * NH_);  // (128, 16, 32)
  scores_kernel<<<gs, 128, 0, stream>>>(qr, qi, kr, ki, attn);
  softmax_kernel<<<B_ * NH_ * S_, 256, 0, stream>>>(attn);
  // 7) ctx = attn @ v (WMMA); 4 waves cover HD=128 in 32-wide slices
  dim3 ga(S_ / 16, B_ * NH_);  // (128, 32)
  attnv_kernel<<<ga, 128, 0, stream>>>(attn, vtr, vti, ctxr, ctxi);
  // 8) complex output projection -> f32 d_out
  cgemm_kernel<false><<<gq, 128, 0, stream>>>(ctxr, ctxi, wort, woit, woin, bo_r, bo_i, outR, outI, HID_, HID_);
}
